// MultiStageGVPModel_64931315580985
// MI455X (gfx1250) — matrix operand order repaired and checked by reading, not compiled, not run on hardware
//
#include <hip/hip_runtime.h>

#define B_    4
#define NL_   128
#define NT_   512
#define NS_   128
#define HD_   256
#define PAIRS_ (B_*NL_*NT_)

typedef __attribute__((ext_vector_type(16))) __bf16 v16bf;
typedef __attribute__((ext_vector_type(8)))  float  v8f;

__device__ __forceinline__ unsigned short f2bf(float f) {
  unsigned u = __float_as_uint(f);
  u += 0x7FFFu + ((u >> 16) & 1u);        // round-to-nearest-even
  return (unsigned short)(u >> 16);
}

__global__ void zero_out_kernel(float* out) {
  if (threadIdx.x < B_ * 4) out[threadIdx.x] = 0.f;
}

// Convert W1 [HD, NS] fp32 -> bf16, pre-swizzled into WMMA B-fragment order:
// tile (kt,nt) -> 32 lanes x 16 bf16; lane = (k&16) + (n&15); slot = k&15.
__global__ void swizzle_weights(const float* __restrict__ W,
                                unsigned short* __restrict__ dst) {
  int i = blockIdx.x * blockDim.x + threadIdx.x;
  if (i >= HD_ * NS_) return;
  int k = i >> 7;            // / NS_
  int n = i & (NS_ - 1);
  unsigned short h = f2bf(W[i]);
  int kt = k >> 5, kk = k & 31;
  int nt = n >> 4, nn = n & 15;
  int lane = (kk & 16) + nn;
  int pos  = kk & 15;
  dst[((((kt << 3) + nt) << 5) + lane) * 16 + pos] = h;
}

// Build a bf16 A-fragment (16x32 tile, one lane's 16 values) from a row of
// fp32 h_cat using native f32->bf16 conversion (v_cvt_pk_bf16_f32 class ops).
__device__ __forceinline__ v16bf make_afrag(const float* __restrict__ hr) {
  float4 x0 = *(const float4*)(hr);
  float4 x1 = *(const float4*)(hr + 4);
  float4 x2 = *(const float4*)(hr + 8);
  float4 x3 = *(const float4*)(hr + 12);
  v16bf v;
  v[ 0] = (__bf16)x0.x; v[ 1] = (__bf16)x0.y; v[ 2] = (__bf16)x0.z; v[ 3] = (__bf16)x0.w;
  v[ 4] = (__bf16)x1.x; v[ 5] = (__bf16)x1.y; v[ 6] = (__bf16)x1.z; v[ 7] = (__bf16)x1.w;
  v[ 8] = (__bf16)x2.x; v[ 9] = (__bf16)x2.y; v[10] = (__bf16)x2.z; v[11] = (__bf16)x2.w;
  v[12] = (__bf16)x3.x; v[13] = (__bf16)x3.y; v[14] = (__bf16)x3.z; v[15] = (__bf16)x3.w;
  return v;
}

__global__ __launch_bounds__(256) void fused_dock_kernel(
    const float* __restrict__ inter,  const float* __restrict__ lpos,
    const float* __restrict__ tpos,   const float* __restrict__ rotor,
    const float* __restrict__ lvdw,   const float* __restrict__ tvdw,
    const float* __restrict__ lnm,    const float* __restrict__ tnm,
    const float* __restrict__ hcat,
    const float* __restrict__ Ab1, const float* __restrict__ AW2, const float* __restrict__ Ab2,
    const float* __restrict__ Bb1, const float* __restrict__ BW2, const float* __restrict__ Bb2,
    const float* __restrict__ hbond_c, const float* __restrict__ hydro_c,
    const float* __restrict__ rotor_c,
    const unsigned short* __restrict__ wsA, const unsigned short* __restrict__ wsB,
    float* __restrict__ out)
{
  const int tid   = threadIdx.x;
  const int w     = tid >> 5;
  const int lane  = tid & 31;
  const int base  = blockIdx.x * 256;          // 256 pairs per block
  const int batch = base / (NL_ * NT_);        // blocks never straddle a batch
  const int rbase = base + w * 32;             // 32 rows per wave (2 strips)

  const v16bf* wAf = (const v16bf*)wsA;
  const v16bf* wBf = (const v16bf*)wsB;

  const v8f zz = {0.f,0.f,0.f,0.f,0.f,0.f,0.f,0.f};
  v8f accA[2][8], accB[2][8];
  #pragma unroll
  for (int s = 0; s < 2; ++s)
    #pragma unroll
    for (int nt = 0; nt < 8; ++nt) { accA[s][nt] = zz; accB[s][nt] = zz; }

  const int mrow  = lane & 15;
  const int khalf = (lane < 16) ? 0 : 16;      // bf16 A-fragment K layout

  // ---- K loop: 8 tiles of K=32, both MLPs share the A (h_cat) fragment ----
  for (int kt = 0; kt < 8; ++kt) {
    v16bf af0 = make_afrag(hcat + (size_t)(rbase +      mrow) * HD_ + kt*32 + khalf);
    v16bf af1 = make_afrag(hcat + (size_t)(rbase + 16 + mrow) * HD_ + kt*32 + khalf);
    #pragma unroll
    for (int nt = 0; nt < 8; ++nt) {
      v16bf wa = wAf[(((kt << 3) + nt) << 5) + lane];
      v16bf wb = wBf[(((kt << 3) + nt) << 5) + lane];
      accA[0][nt] = __builtin_amdgcn_wmma_f32_16x16x32_bf16(
          false, af0, false, wa, (short)0, accA[0][nt], false, false);
      accB[0][nt] = __builtin_amdgcn_wmma_f32_16x16x32_bf16(
          false, af0, false, wb, (short)0, accB[0][nt], false, false);
      accA[1][nt] = __builtin_amdgcn_wmma_f32_16x16x32_bf16(
          false, af1, false, wa, (short)0, accA[1][nt], false, false);
      accB[1][nt] = __builtin_amdgcn_wmma_f32_16x16x32_bf16(
          false, af1, false, wb, (short)0, accB[1][nt], false, false);
    }
  }

  // ---- second layer constants (N = nt*16 + lane%16 per C-layout) ----
  float b1A[8], b1B[8], w2A[8], w2B[8];
  #pragma unroll
  for (int nt = 0; nt < 8; ++nt) {
    int N = nt * 16 + mrow;
    b1A[nt] = Ab1[N]; b1B[nt] = Bb1[N];
    w2A[nt] = AW2[N]; w2B[nt] = BW2[N];
  }
  const float b2A = Ab2[0], b2B = Bb2[0];

  float e_vdw = 0.f, e_hb = 0.f, e_me = 0.f, e_hy = 0.f;

  for (int s = 0; s < 2; ++s) {
    // per-row dot products: relu(hid + b1) @ W2, reduced across the 16 lanes
    float dA = 0.f, dB = 0.f;    // lanes 0..15 end up holding row = lane
    #pragma unroll
    for (int j = 0; j < 8; ++j) {
      float tA = 0.f, tB = 0.f;
      #pragma unroll
      for (int nt = 0; nt < 8; ++nt) {
        float ra = fmaxf(accA[s][nt][j] + b1A[nt], 0.f);
        float rb = fmaxf(accB[s][nt][j] + b1B[nt], 0.f);
        tA = fmaf(ra, w2A[nt], tA);
        tB = fmaf(rb, w2B[nt], tB);
      }
      #pragma unroll
      for (int m = 1; m <= 8; m <<= 1) {      // butterfly within each 16-lane half
        tA += __shfl_xor(tA, m, 32);
        tB += __shfl_xor(tB, m, 32);
      }
      float loA = __shfl(tA, 0, 32),  hiA = __shfl(tA, 16, 32);
      float loB = __shfl(tB, 0, 32),  hiB = __shfl(tB, 16, 32);
      if (lane == j)     { dA = loA; dB = loB; }   // row j
      if (lane == j + 8) { dA = hiA; dB = hiB; }   // row j+8
    }

    if (lane < 16) {
      int p = rbase + s * 16 + lane;
      int t = p & (NT_ - 1);
      int l = (p >> 9) & (NL_ - 1);

      float Aw = 1.f / (1.f + __expf(-(dA + b2A)));          // sigmoid
      Aw = Aw * (0.0356f - 0.0178f) + 0.0178f;
      float Bp = tanhf(dB + b2B) * 0.2f;                     // DEV_VDW_RADIUS

      const float* lp = lpos + ((size_t)batch * NL_ + l) * 3;
      const float* tp = tpos + ((size_t)batch * NT_ + t) * 3;
      float dx = lp[0]-tp[0], dy = lp[1]-tp[1], dz = lp[2]-tp[2];
      float dm = sqrtf(dx*dx + dy*dy + dz*dz + 1e-10f);
      if (dm < 0.5f) dm = 1e10f;

      float dm0 = lvdw[batch*NL_ + l] + tvdw[batch*NT_ + t] + Bp;
      float ddm = dm - dm0;

      size_t ibase = (((size_t)batch * 3) * NL_ + l) * NT_ + t;
      float i0 = inter[ibase];
      float i1 = inter[ibase + (size_t)NL_ * NT_];
      float i2 = inter[ibase + 2 * (size_t)NL_ * NT_];

      e_hb += fminf(fmaxf(ddm * i0 * (-1.f / 0.7f), 0.f), 1.f);
      e_me += fminf(fmaxf(ddm * i1 * (-1.f / 0.7f), 0.f), 1.f);
      e_hy += fminf(fmaxf((1.5f - ddm) * i2, 0.f), 1.f);

      float dm0v = (dm0 < 1e-4f) ? 1.f : dm0;
      float r  = dm0v / dm;
      float r2 = r * r, r6 = r2 * r2 * r2, r12 = r6 * r6;
      float ve = fminf(r12 - 2.f * r6, 100.f);
      ve *= lnm[batch*NL_ + l] * tnm[batch*NT_ + t];
      e_vdw += Aw * ve;
    }
  }

  // ---- wave reduction (lanes >=16 carry zeros) ----
  #pragma unroll
  for (int m = 1; m <= 16; m <<= 1) {
    e_vdw += __shfl_xor(e_vdw, m, 32);
    e_hb  += __shfl_xor(e_hb,  m, 32);
    e_me  += __shfl_xor(e_me,  m, 32);
    e_hy  += __shfl_xor(e_hy,  m, 32);
  }

  __shared__ float ered[8][4];
  if (lane == 0) {
    float hc = hbond_c[0], hp = hydro_c[0], rc = rotor_c[0];
    float scale = 1.f / (1.f + rc * rc * rotor[batch]);   // linear -> fold in now
    ered[w][0] = e_vdw * scale;
    ered[w][1] = -(hc * hc) * e_hb * scale;
    ered[w][2] = -(hc * hc) * e_me * scale;
    ered[w][3] = -(hp * hp) * e_hy * scale;
  }
  __syncthreads();
  if (tid < 4) {
    float v = 0.f;
    #pragma unroll
    for (int i = 0; i < 8; ++i) v += ered[i][tid];
    atomicAdd(out + batch * 4 + tid, v);
  }
}

extern "C" void kernel_launch(void* const* d_in, const int* in_sizes, int n_in,
                              void* d_out, int out_size, void* d_ws, size_t ws_size,
                              hipStream_t stream) {
  const float* inter   = (const float*)d_in[0];
  const float* lpos    = (const float*)d_in[1];
  const float* tpos    = (const float*)d_in[2];
  const float* rotor   = (const float*)d_in[3];
  const float* lvdw    = (const float*)d_in[4];
  const float* tvdw    = (const float*)d_in[5];
  const float* lnm     = (const float*)d_in[6];
  const float* tnm     = (const float*)d_in[7];
  const float* hcat    = (const float*)d_in[8];
  const float* AW1     = (const float*)d_in[9];
  const float* Ab1     = (const float*)d_in[10];
  const float* AW2     = (const float*)d_in[11];
  const float* Ab2     = (const float*)d_in[12];
  const float* BW1     = (const float*)d_in[13];
  const float* Bb1     = (const float*)d_in[14];
  const float* BW2     = (const float*)d_in[15];
  const float* Bb2     = (const float*)d_in[16];
  const float* hbond_c = (const float*)d_in[17];
  const float* hydro_c = (const float*)d_in[18];
  const float* rotor_c = (const float*)d_in[19];

  unsigned short* wsA = (unsigned short*)d_ws;        // 64 KB swizzled bf16
  unsigned short* wsB = wsA + HD_ * NS_;              // next 64 KB

  zero_out_kernel<<<1, 64, 0, stream>>>((float*)d_out);
  swizzle_weights<<<(HD_ * NS_ + 255) / 256, 256, 0, stream>>>(AW1, wsA);
  swizzle_weights<<<(HD_ * NS_ + 255) / 256, 256, 0, stream>>>(BW1, wsB);
  fused_dock_kernel<<<PAIRS_ / 256, 256, 0, stream>>>(
      inter, lpos, tpos, rotor, lvdw, tvdw, lnm, tnm, hcat,
      Ab1, AW2, Ab2, Bb1, BW2, Bb2, hbond_c, hydro_c, rotor_c,
      wsA, wsB, (float*)d_out);
}
